// BDH_52209622450688
// MI455X (gfx1250) — compile-verified, matrix operand order checked by reading, not run on hardware
//
#include <hip/hip_runtime.h>

#define DEV __device__ __forceinline__

typedef __attribute__((ext_vector_type(16))) __bf16 v16bf;
typedef __attribute__((ext_vector_type(8)))  __bf16 v8bf;
typedef __attribute__((ext_vector_type(8)))  float  v8f;

constexpr int   T_   = 2048;
constexpr int   D_   = 256;
constexpr int   NH_  = 4;
constexpr int   NN_  = 8192;        // N per head
constexpr int   KNN_ = NH_ * NN_;   // 32768
constexpr float EPS_ = 1e-5f;

// ---------------------------------------------------------- async-LDS path ---
#if defined(__has_builtin)
# if __has_builtin(__builtin_amdgcn_global_load_async_to_lds_b128)
#  define USE_ASYNC_LDS 1
# endif
#endif
#ifndef USE_ASYNC_LDS
# define USE_ASYNC_LDS 0
#endif

// Exact pointee type from the hipcc diagnostic: vector_size(16) int.
typedef int v4i_ __attribute__((vector_size(16)));
typedef __attribute__((address_space(1))) v4i_ as1_v4i;
typedef __attribute__((address_space(3))) v4i_ as3_v4i;

DEV as1_v4i* as_g128(const void* p) {
  // global flat address == as1 address
  return (as1_v4i*)(unsigned long long)(uintptr_t)p;
}
DEV as3_v4i* as_l128(void* p) {
  // ISA 10.2: LDS aperture -> LDS address is the low 32 bits
  return (as3_v4i*)(unsigned)(uintptr_t)p;
}

#if USE_ASYNC_LDS
# if __has_builtin(__builtin_amdgcn_s_wait_asynccnt)
#  define WAIT_ASYNC(n) __builtin_amdgcn_s_wait_asynccnt(n)
# else
#  define WAIT_ASYNC(n) asm volatile("s_wait_asynccnt " #n ::: "memory")
# endif
#else
# define WAIT_ASYNC(n) ((void)0)
#endif

// ------------------------------------------------------------ WMMA helpers ---
union FragU { v16bf v; v8bf h[2]; };

// 16x32 bf16 fragment; ISA layout: lane&15 = row, lanes 0-15 hold
// K {0..7,16..23}, lanes 16-31 hold K {8..15,24..31} -> two 16B loads.
DEV v16bf load_frag(const __bf16* base, int stride, int lane) {
  int m  = lane & 15;
  int hi = lane >> 4;
  const __bf16* p = base + (size_t)m * stride + hi * 8;
  FragU f;
  f.h[0] = *reinterpret_cast<const v8bf*>(p);
  f.h[1] = *reinterpret_cast<const v8bf*>(p + 16);
  return f.v;
}

DEV v8f wmma_bf16(v16bf a, v16bf b, v8f c) {
  return __builtin_amdgcn_wmma_f32_16x16x32_bf16(false, a, false, b,
                                                 (short)0, c, false, false);
}

DEV v8f vzero8() {
  v8f z = {0.f, 0.f, 0.f, 0.f, 0.f, 0.f, 0.f, 0.f};
  return z;
}

DEV float block_sum(float v, float* sm) {
  int tid = threadIdx.x;
  sm[tid] = v; __syncthreads();
  for (int s = 128; s > 0; s >>= 1) {
    if (tid < s) sm[tid] += sm[tid + s];
    __syncthreads();
  }
  float r = sm[0];
  __syncthreads();
  return r;
}

// ---------------------------------------------------------------- weights ----
__global__ void transpose_f32_bf16_kernel(const float* __restrict__ src,
                                          __bf16* __restrict__ dst,
                                          int R, int C) {
  size_t i = (size_t)blockIdx.x * blockDim.x + threadIdx.x;
  if (i >= (size_t)R * C) return;
  int r = (int)(i / C), c = (int)(i % C);
  dst[(size_t)c * R + r] = (__bf16)src[i];
}

// ------------------------------------------------------------- embed + LN ----
__global__ __launch_bounds__(256) void embed_ln_kernel(
    const int* __restrict__ idx, const float* __restrict__ embed,
    float* __restrict__ xf, __bf16* __restrict__ xb, __bf16* __restrict__ xTb) {
  __shared__ float sm[256];
  int t = blockIdx.x, d = threadIdx.x;
  float e   = embed[(size_t)idx[t] * D_ + d];
  float mu  = block_sum(e, sm) * (1.f / D_);
  float df  = e - mu;
  float var = block_sum(df * df, sm) * (1.f / D_);
  float o   = df * rsqrtf(var + EPS_);
  xf [(size_t)t * D_ + d] = o;
  xb [(size_t)t * D_ + d] = (__bf16)o;
  xTb[(size_t)d * T_ + t] = (__bf16)o;
}

// --------------------------------------------- x_sparse = relu(x@enc), RoPE --
__global__ __launch_bounds__(256) void sparse_rope_kernel(
    const __bf16* __restrict__ xb, const __bf16* __restrict__ encT,
    __bf16* __restrict__ xs, __bf16* __restrict__ qr) {
  int lane = threadIdx.x & 31, wave = threadIdx.x >> 5;
  int wm = wave >> 1, wn = wave & 1;
  int h  = blockIdx.z;
  int t0 = blockIdx.y * 64 + wm * 16;
  int n0 = blockIdx.x * 64 + wn * 32;
  const __bf16* A  = xb + (size_t)t0 * D_;
  const __bf16* Bt = encT + (size_t)h * NN_ * D_;
  v8f c0 = vzero8(), c1 = vzero8();
#pragma unroll
  for (int k = 0; k < D_; k += 32) {
    v16bf a  = load_frag(A + k, D_, lane);
    v16bf b0 = load_frag(Bt + (size_t)n0 * D_ + k, D_, lane);
    v16bf b1 = load_frag(Bt + (size_t)(n0 + 16) * D_ + k, D_, lane);
    c0 = wmma_bf16(a, b0, c0);
    c1 = wmma_bf16(a, b1, c1);
  }
  int col = lane & 15, hi = lane >> 4;
  __bf16* xsH = xs + (size_t)h * T_ * NN_;
  __bf16* qrH = qr + (size_t)h * T_ * NN_;
#pragma unroll
  for (int j = 0; j < 2; ++j) {
#pragma unroll
    for (int r = 0; r < 8; ++r) {
      int tt = t0 + r + 8 * hi;
      int nn = n0 + j * 16 + col;
      float v = (j == 0) ? c0[r] : c1[r];
      v = v > 0.f ? v : 0.f;                       // relu -> x_sparse
      xsH[(size_t)tt * NN_ + nn] = (__bf16)v;
      // RoPE pair (even,odd) sits in adjacent lanes of the C layout.
      float partner = __shfl_xor(v, 1, 32);
      float rot = (nn & 1) ? partner : -partner;   // (-v[n+1], v[n])
      float fr = exp2f((float)(nn & ~1) * (-16.0f / (float)NN_))
               * 0.15915494309189535f;             // THETA=2^16, /(2pi)
      float ph = (float)tt * fr;
      ph -= floorf(ph);
      float sn, cs;
      sincosf(ph * 6.283185307179586f, &sn, &cs);
      qrH[(size_t)tt * NN_ + nn] = (__bf16)(v * cs + rot * sn);
    }
  }
}

// ------------------------------- yKV = (mask * (QR @ QR^T)) @ x  (flash) ----
// Double-buffered async k-slabs of 128 over K=8192; dynamic LDS.
constexpr int    KSLAB   = 128;
constexpr int    APITCH  = 136;                     // 128 + 8 pad (68 dwords)
constexpr size_t ATTN_LDS_BYTES =
    (size_t)2 * 2 * 64 * APITCH * 2 /* A+B double buffered */ +
    (size_t)64 * 72 * 2 /* S */;

// Each thread stages 4x b128 per matrix per slab (8 async ops per slab).
DEV void stage_slab(const __bf16* gA, const __bf16* gB,
                    __bf16* lA, __bf16* lB) {
#pragma unroll
  for (int j = 0; j < 4; ++j) {
#if USE_ASYNC_LDS
    __builtin_amdgcn_global_load_async_to_lds_b128(
        as_g128(gA + j * 8), as_l128(lA + j * 8), 0, 0);
    __builtin_amdgcn_global_load_async_to_lds_b128(
        as_g128(gB + j * 8), as_l128(lB + j * 8), 0, 0);
#else
    *reinterpret_cast<v8bf*>(lA + j * 8) =
        *reinterpret_cast<const v8bf*>(gA + j * 8);
    *reinterpret_cast<v8bf*>(lB + j * 8) =
        *reinterpret_cast<const v8bf*>(gB + j * 8);
#endif
  }
}

__global__ __launch_bounds__(256) void attn_kernel(
    const __bf16* __restrict__ qr, const __bf16* __restrict__ xTb,
    float* __restrict__ ykv) {
  extern __shared__ __align__(16) char smem[];
  __bf16* ldsA = (__bf16*)smem;                 // [2][64][APITCH]
  __bf16* ldsB = ldsA + 2 * 64 * APITCH;        // [2][64][APITCH]
  __bf16* ldsS = ldsB + 2 * 64 * APITCH;        // [64][72]
  int lane = threadIdx.x & 31, wave = threadIdx.x >> 5;
  int wm = wave >> 1, wn = wave & 1;
  int h  = blockIdx.y;
  int t0 = blockIdx.x * 64;
  const __bf16* Q = qr + (size_t)h * T_ * NN_;
  int col = lane & 15, hi = lane >> 4;
  v8f acc[8];
#pragma unroll
  for (int j = 0; j < 8; ++j) acc[j] = vzero8();
  int srow = threadIdx.x >> 2;                  // 0..63 staging row
  int scol = (threadIdx.x & 3) * 32;            // 32 bf16 per matrix chunk
  const __bf16* gArow = Q + (size_t)(t0 + srow) * NN_ + scol;
  constexpr int NSLAB = NN_ / KSLAB;            // 64
  for (int s0 = 0; s0 <= t0; s0 += 64) {
    const __bf16* gBrow = Q + (size_t)(s0 + srow) * NN_ + scol;
    v8f sc0 = vzero8(), sc1 = vzero8();
    // prologue: slab 0 -> buffer 0
    stage_slab(gArow, gBrow,
               ldsA + (size_t)srow * APITCH + scol,
               ldsB + (size_t)srow * APITCH + scol);
    for (int i = 0; i < NSLAB; ++i) {
      int cur = i & 1;
      if (i + 1 < NSLAB) {
        int nb = 1 - cur;
        stage_slab(gArow + (i + 1) * KSLAB, gBrow + (i + 1) * KSLAB,
                   ldsA + ((size_t)nb * 64 + srow) * APITCH + scol,
                   ldsB + ((size_t)nb * 64 + srow) * APITCH + scol);
        WAIT_ASYNC(8);     // in-order: previous slab's 8 ops complete
      } else {
        WAIT_ASYNC(0);
      }
      __syncthreads();
      const __bf16* bufA = ldsA + (size_t)cur * 64 * APITCH;
      const __bf16* bufB = ldsB + (size_t)cur * 64 * APITCH;
#pragma unroll
      for (int kk = 0; kk < KSLAB; kk += 32) {
        v16bf a  = load_frag(bufA + (size_t)(wm * 16) * APITCH + kk, APITCH, lane);
        v16bf b0 = load_frag(bufB + (size_t)(wn * 32) * APITCH + kk, APITCH, lane);
        v16bf b1 = load_frag(bufB + (size_t)(wn * 32 + 16) * APITCH + kk, APITCH, lane);
        sc0 = wmma_bf16(a, b0, sc0);
        sc1 = wmma_bf16(a, b1, sc1);
      }
      __syncthreads();
    }
    // causal mask (keep t > s), stash S as bf16 in LDS
#pragma unroll
    for (int rr = 0; rr < 8; ++rr) {
      int lt = wm * 16 + rr + 8 * hi;
      int gt = t0 + lt;
      int ls0 = wn * 32 + col;
      ldsS[lt * 72 + ls0] = (__bf16)((gt > s0 + ls0) ? sc0[rr] : 0.f);
      int ls1 = ls0 + 16;
      ldsS[lt * 72 + ls1] = (__bf16)((gt > s0 + ls1) ? sc1[rr] : 0.f);
    }
    __syncthreads();
    // yKV += S @ x_tile ; wave owns 16 rows x 128 cols of yKV
#pragma unroll
    for (int kk = 0; kk < 64; kk += 32) {
      v16bf a = load_frag(ldsS + (size_t)(wm * 16) * 72 + kk, 72, lane);
#pragma unroll
      for (int j = 0; j < 8; ++j) {
        int d0 = wn * 128 + j * 16;
        v16bf b = load_frag(xTb + (size_t)d0 * T_ + s0 + kk, T_, lane);
        acc[j] = wmma_bf16(a, b, acc[j]);
      }
    }
    __syncthreads();
  }
  float* Y = ykv + ((size_t)h * T_ + t0 + wm * 16) * D_;
#pragma unroll
  for (int j = 0; j < 8; ++j)
#pragma unroll
    for (int rr = 0; rr < 8; ++rr) {
      int lr = rr + 8 * hi;
      Y[(size_t)lr * D_ + wn * 128 + j * 16 + col] = acc[j][rr];
    }
}

// -------------------------------------------------- LN rows (f32 -> bf16) ----
__global__ __launch_bounds__(256) void ln_rows_kernel(
    const float* __restrict__ in, __bf16* __restrict__ out) {
  __shared__ float sm[256];
  size_t row = blockIdx.x;
  int d = threadIdx.x;
  float v   = in[row * D_ + d];
  float mu  = block_sum(v, sm) * (1.f / D_);
  float df  = v - mu;
  float var = block_sum(df * df, sm) * (1.f / D_);
  out[row * D_ + d] = (__bf16)(df * rsqrtf(var + EPS_));
}

// ------------------------------- xy = relu(ln(yKV) @ encV) * x_sparse -------
__global__ __launch_bounds__(256) void ys_xy_kernel(
    const __bf16* __restrict__ ykn, const __bf16* __restrict__ encVT,
    const __bf16* __restrict__ xs, __bf16* __restrict__ xy) {
  int lane = threadIdx.x & 31, wave = threadIdx.x >> 5;
  int wm = wave >> 1, wn = wave & 1;
  int h  = blockIdx.z;
  int t0 = blockIdx.y * 64 + wm * 16;
  int n0 = blockIdx.x * 64 + wn * 32;
  const __bf16* A  = ykn + ((size_t)h * T_ + t0) * D_;
  const __bf16* Bt = encVT + (size_t)h * NN_ * D_;
  v8f c0 = vzero8(), c1 = vzero8();
#pragma unroll
  for (int k = 0; k < D_; k += 32) {
    v16bf a  = load_frag(A + k, D_, lane);
    v16bf b0 = load_frag(Bt + (size_t)n0 * D_ + k, D_, lane);
    v16bf b1 = load_frag(Bt + (size_t)(n0 + 16) * D_ + k, D_, lane);
    c0 = wmma_bf16(a, b0, c0);
    c1 = wmma_bf16(a, b1, c1);
  }
  int col = lane & 15, hi = lane >> 4;
  const __bf16* xsH = xs + (size_t)h * T_ * NN_;
#pragma unroll
  for (int j = 0; j < 2; ++j)
#pragma unroll
    for (int r = 0; r < 8; ++r) {
      int tt = t0 + r + 8 * hi;
      int nn = n0 + j * 16 + col;
      float v = (j == 0) ? c0[r] : c1[r];
      v = v > 0.f ? v : 0.f;
      float xsv = (float)xsH[(size_t)tt * NN_ + nn];
      xy[(size_t)tt * KNN_ + (size_t)h * NN_ + nn] = (__bf16)(v * xsv);
    }
}

// ------------------------------------------------ yMLP = xy @ decoder --------
__global__ __launch_bounds__(256) void mlp_kernel(
    const __bf16* __restrict__ xy, const __bf16* __restrict__ decT,
    float* __restrict__ ymlp) {
  int lane = threadIdx.x & 31, wave = threadIdx.x >> 5;
  int wm = wave >> 1, wn = wave & 1;
  int t0 = blockIdx.y * 64 + wm * 16;
  int d0 = blockIdx.x * 64 + wn * 32;
  const __bf16* Arow = xy + (size_t)t0 * KNN_;
  const __bf16* B0   = decT + (size_t)d0 * KNN_;
  const __bf16* B1   = decT + (size_t)(d0 + 16) * KNN_;
  v8f c0 = vzero8(), c1 = vzero8();
  for (int k = 0; k < KNN_; k += 32) {
    if ((k & 255) == 0 && k + 2048 < KNN_) {   // pull cold HBM lines early
      __builtin_prefetch(Arow + k + 2048, 0, 1);
      __builtin_prefetch(B0 + k + 2048, 0, 1);
      __builtin_prefetch(B1 + k + 2048, 0, 1);
    }
    v16bf a  = load_frag(Arow + k, KNN_, lane);
    v16bf b0 = load_frag(B0 + k, KNN_, lane);
    v16bf b1 = load_frag(B1 + k, KNN_, lane);
    c0 = wmma_bf16(a, b0, c0);
    c1 = wmma_bf16(a, b1, c1);
  }
  int col = lane & 15, hi = lane >> 4;
#pragma unroll
  for (int j = 0; j < 2; ++j)
#pragma unroll
    for (int r = 0; r < 8; ++r) {
      int tt = t0 + r + 8 * hi;
      int dd = d0 + j * 16 + col;
      ymlp[(size_t)tt * D_ + dd] = (j == 0) ? c0[r] : c1[r];
    }
}

// --------------------------------------------- x = ln(x + ln(yMLP)) ----------
__global__ __launch_bounds__(256) void add_ln_kernel(
    const float* __restrict__ ymlp, float* __restrict__ xf,
    __bf16* __restrict__ xb, __bf16* __restrict__ xTb) {
  __shared__ float sm[256];
  int t = blockIdx.x, d = threadIdx.x;
  float v    = ymlp[(size_t)t * D_ + d];
  float mu   = block_sum(v, sm) * (1.f / D_);
  float df   = v - mu;
  float var  = block_sum(df * df, sm) * (1.f / D_);
  float l1   = df * rsqrtf(var + EPS_);
  float u    = xf[(size_t)t * D_ + d] + l1;
  float mu2  = block_sum(u, sm) * (1.f / D_);
  float df2  = u - mu2;
  float var2 = block_sum(df2 * df2, sm) * (1.f / D_);
  float o    = df2 * rsqrtf(var2 + EPS_);
  xf [(size_t)t * D_ + d] = o;
  xb [(size_t)t * D_ + d] = (__bf16)o;
  xTb[(size_t)d * T_ + t] = (__bf16)o;
}

// ------------------------------------------------ logits = x @ lm_head -------
__global__ __launch_bounds__(256) void lm_head_kernel(
    const float* __restrict__ xf, const float* __restrict__ W,
    float* __restrict__ out) {
  __shared__ float xr[256];
  int t = blockIdx.x, v = threadIdx.x;
  xr[v] = xf[(size_t)t * D_ + v];
  __syncthreads();
  float s = 0.f;
#pragma unroll 8
  for (int d = 0; d < D_; ++d) s = fmaf(xr[d], W[(size_t)d * 256 + v], s);
  out[(size_t)t * 256 + v] = s;
}

// ---------------------------------------------------------------- launcher ---
extern "C" void kernel_launch(void* const* d_in, const int* in_sizes, int n_in,
                              void* d_out, int out_size, void* d_ws,
                              size_t ws_size, hipStream_t stream) {
  const int*   idx      = (const int*)  d_in[0];
  const float* embed    = (const float*)d_in[1];
  const float* encoder  = (const float*)d_in[2];
  const float* encoderv = (const float*)d_in[3];
  const float* decoder  = (const float*)d_in[4];
  const float* lmhead   = (const float*)d_in[5];
  float* out = (float*)d_out;

  char* p = (char*)d_ws;
  auto alloc = [&](size_t bytes) -> char* {
    char* r = p;
    p += (bytes + 255) & ~(size_t)255;
    return r;
  };
  float*  xf    = (float*) alloc((size_t)T_ * D_ * 4);
  __bf16* xb    = (__bf16*)alloc((size_t)T_ * D_ * 2);
  __bf16* xTb   = (__bf16*)alloc((size_t)D_ * T_ * 2);
  __bf16* encT  = (__bf16*)alloc((size_t)NH_ * NN_ * D_ * 2);
  __bf16* encVT = (__bf16*)alloc((size_t)NH_ * NN_ * D_ * 2);
  __bf16* decT  = (__bf16*)alloc((size_t)D_ * KNN_ * 2);
  __bf16* xs    = (__bf16*)alloc((size_t)NH_ * T_ * NN_ * 2);
  __bf16* qrxy  = (__bf16*)alloc((size_t)NH_ * T_ * NN_ * 2); // QR, reused as xy
  float*  ykv   = (float*) alloc((size_t)NH_ * T_ * D_ * 4);
  __bf16* ykn   = (__bf16*)alloc((size_t)NH_ * T_ * D_ * 2);
  float*  ymlp  = (float*) alloc((size_t)T_ * D_ * 4);

  for (int h = 0; h < NH_; ++h) {
    transpose_f32_bf16_kernel<<<(D_ * NN_ + 255) / 256, 256, 0, stream>>>(
        encoder + (size_t)h * D_ * NN_, encT + (size_t)h * NN_ * D_, D_, NN_);
    transpose_f32_bf16_kernel<<<(D_ * NN_ + 255) / 256, 256, 0, stream>>>(
        encoderv + (size_t)h * D_ * NN_, encVT + (size_t)h * NN_ * D_, D_, NN_);
  }
  transpose_f32_bf16_kernel<<<(KNN_ * D_ + 255) / 256, 256, 0, stream>>>(
      decoder, decT, KNN_, D_);

  embed_ln_kernel<<<T_, 256, 0, stream>>>(idx, embed, xf, xb, xTb);

  for (int layer = 0; layer < 2; ++layer) {
    sparse_rope_kernel<<<dim3(NN_ / 64, T_ / 64, NH_), 256, 0, stream>>>(
        xb, encT, xs, qrxy);
    attn_kernel<<<dim3(T_ / 64, NH_), 256, (unsigned)ATTN_LDS_BYTES, stream>>>(
        qrxy, xTb, ykv);
    ln_rows_kernel<<<NH_ * T_, 256, 0, stream>>>(ykv, ykn);
    ys_xy_kernel<<<dim3(NN_ / 64, T_ / 64, NH_), 256, 0, stream>>>(
        ykn, encVT, xs, qrxy);   // QR dead -> reuse buffer for xy
    mlp_kernel<<<dim3(D_ / 64, T_ / 64), 256, 0, stream>>>(qrxy, decT, ymlp);
    add_ln_kernel<<<T_, 256, 0, stream>>>(ymlp, xf, xb, xTb);
  }
  lm_head_kernel<<<T_, 256, 0, stream>>>(xf, lmhead, out);
}
